// RandomAttention_77464030151330
// MI455X (gfx1250) — compile-verified, matrix operand order checked by reading, not run on hardware
//
#include <hip/hip_runtime.h>
#include <hip/hip_bf16.h>
#include <stdint.h>

#define S_LEN 2048
#define D_MODEL 768
#define NB 4
#define NH 12
#define DHEAD 64

typedef __attribute__((ext_vector_type(16))) __bf16 v16bf;
typedef __attribute__((ext_vector_type(8)))  float  v8f;

union Frag {
  uint32_t u[8];
  v16bf    v;
};

__device__ __forceinline__ uint16_t f2bf(float f) {
  union { float f; uint32_t u; } x; x.f = f;
  uint32_t u = x.u;
  u += 0x7fffu + ((u >> 16) & 1u);   // round-to-nearest-even
  return (uint16_t)(u >> 16);
}
__device__ __forceinline__ uint32_t pack2(float a, float b) {
  return (uint32_t)f2bf(a) | ((uint32_t)f2bf(b) << 16);
}

// The TDM builtin carries no pointer operand, so LLVM would otherwise treat
// the LDS buffer as never-written and fold our ds_loads to undef.  Escaping
// the pointer into an asm block with a memory clobber makes the compiler
// assume the DMA wrote it.
__device__ __forceinline__ void lds_written_fence(const void* p) {
  asm volatile("" : : "r"(p) : "memory");
}

// Gather one 16x32 bf16 A/B-layout fragment for (row, k0..k0+31) from a
// row-major bf16 matrix (global or LDS) with leading dim `ld`.
// Layout (ISA 7.12.2, 16-bit A 16x32): half = lane>>4;
//   VGPR i<4 : K = k0 + 8*half + 2i,2i+1
//   VGPR i>=4: K = k0 + 16 + 8*half + 2(i-4),+1
__device__ __forceinline__ void load_frag(const uint16_t* base,
                                          int row, int ld, int k0, Frag& f) {
  const int lane = threadIdx.x & 31;
  const int half = lane >> 4;
  const uint32_t* p = (const uint32_t*)(base + (size_t)row * ld);
  const int ka = (k0 + half * 8) >> 1;
  const int kb = (k0 + 16 + half * 8) >> 1;
#pragma unroll
  for (int i = 0; i < 4; ++i) f.u[i] = p[ka + i];
#pragma unroll
  for (int i = 0; i < 4; ++i) f.u[4 + i] = p[kb + i];
}

__device__ __forceinline__ v8f wmma_bf16(const Frag& a, const Frag& b, v8f c) {
  return __builtin_amdgcn_wmma_f32_16x16x32_bf16(
      /*neg_a=*/false, a.v, /*neg_b=*/false, b.v,
      /*c_mod=*/(short)0, c, /*reuse_a=*/false, /*reuse_b=*/false);
}

// ------------------------------------------------- TDM 2-D tile -> LDS ----
// Builds a Tensor DMA Descriptor (CDNA5 ISA) for a 2-D bf16
// tile: tile0 elems contiguous (x), tile1 rows (y) with stride0 elems between
// rows.  LDS receives the tile packed row-after-row at byte offset lds_off.
// This toolchain's builtin takes 6 args: (g0, g1, g2, g3, g_extra, cpol).
__device__ __forceinline__ void tdm_load_2d(uint32_t lds_off, const void* gptr,
                                            uint32_t tile0, uint32_t tile1,
                                            uint32_t stride0) {
  typedef uint32_t u32x4 __attribute__((ext_vector_type(4)));
  typedef int      i32x8 __attribute__((ext_vector_type(8)));
  typedef int      i32x4 __attribute__((ext_vector_type(4)));
  const uint64_t ga = (uint64_t)(uintptr_t)gptr;
  u32x4 g0;
  g0[0] = 1u;                                  // count=1, no gather
  g0[1] = lds_off;                             // lds_addr (bytes)
  g0[2] = (uint32_t)ga;                        // global_addr[31:0]
  g0[3] = (uint32_t)(ga >> 32) | 0x80000000u;  // addr[56:32] | type=2
  const uint32_t td0 = tile0, td1 = tile1;     // tensor dims == tile dims
  i32x8 g1;
  g1[0] = (int)(1u << 16);                     // data_size = 2 bytes
  g1[1] = (int)((td0 & 0xFFFFu) << 16);        // tensor_dim0 lo
  g1[2] = (int)((td0 >> 16) | ((td1 & 0xFFFFu) << 16));
  g1[3] = (int)((td1 >> 16) | (tile0 << 16));  // tile_dim0
  g1[4] = (int)tile1;                          // tile_dim1 (tile_dim2=0)
  g1[5] = (int)stride0;                        // tensor_dim0_stride lo
  g1[6] = 0;
  g1[7] = 0;
  const i32x4 z4 = {0, 0, 0, 0};               // groups 2/3: 2-D tensor
  const i32x8 z8 = {0, 0, 0, 0, 0, 0, 0, 0};
  __builtin_amdgcn_tensor_load_to_lds(g0, g1, z4, z4, z8, 0);
}

// ---------------------------------------------------------------- prep ----
__global__ void ra_cvt_bf16(const float* __restrict__ x,
                            uint16_t* __restrict__ y, int n) {
  int i = blockIdx.x * blockDim.x + threadIdx.x;
  if (i < n) y[i] = f2bf(x[i]);
}

__global__ void ra_transpose_w(const float* __restrict__ W,
                               uint16_t* __restrict__ Wt) {
  int idx = blockIdx.x * blockDim.x + threadIdx.x;
  if (idx >= D_MODEL * D_MODEL) return;
  int n = idx / D_MODEL, k = idx % D_MODEL;
  Wt[idx] = f2bf(W[(size_t)k * D_MODEL + n]);   // Wt[n][k] = W[k][n]
}

// ------------------------------------------------------------ QKV GEMM ----
// grid = (64, 12, 3), block = 256 (8 waves).  Block: 128 rows x 64 cols.
// Weight tile (64 cols x 32 k, 4KB) TDM-staged in LDS, double-buffered.
__global__ __launch_bounds__(256)
void ra_qkv_gemm(const uint16_t* __restrict__ Xb,
                 const uint16_t* __restrict__ Wqt,
                 const uint16_t* __restrict__ Wkt,
                 const uint16_t* __restrict__ Wvt,
                 const float* __restrict__ bq,
                 const float* __restrict__ bk,
                 const float* __restrict__ bv,
                 uint16_t* __restrict__ Qb,
                 uint16_t* __restrict__ Kb,
                 uint16_t* __restrict__ Vt) {
  __shared__ __align__(16) char smem[8192];     // Bt[2][64*32] bf16

  const int z = blockIdx.z;                     // 0=Q 1=K 2=V
  const uint16_t* Wt   = (z == 0) ? Wqt : (z == 1) ? Wkt : Wvt;
  const float*    bias = (z == 0) ? bq  : (z == 1) ? bk  : bv;

  const int tid  = threadIdx.x;
  const int lane = tid & 31;
  const int wave = tid >> 5;
  const int half = lane >> 4;
  const int rowbase = (blockIdx.x * 8 + wave) * 16;
  const int colbase = blockIdx.y * 64;
  const int arow = rowbase + (lane & 15);

  const uint16_t* wbase = Wt + (size_t)colbase * D_MODEL;  // [col][k]

  if (wave == 0) {                              // stage k-chunk 0 -> buf 0
    tdm_load_2d(0u, wbase, 32, 64, D_MODEL);
    __builtin_amdgcn_s_wait_tensorcnt(0);
  }
  __syncthreads();
  lds_written_fence(smem);

  v8f acc[4] = {v8f{}, v8f{}, v8f{}, v8f{}};
  const int NK = D_MODEL / 32;                  // 24
  for (int ki = 0; ki < NK; ++ki) {
    const int cur = ki & 1;
    if (wave == 0 && ki + 1 < NK)               // DMA next chunk -> other buf
      tdm_load_2d((uint32_t)((cur ^ 1) * 4096), wbase + (ki + 1) * 32,
                  32, 64, D_MODEL);

    const uint16_t* Bt = (const uint16_t*)(smem + cur * 4096);
    Frag a;
    load_frag(Xb, arow, D_MODEL, ki * 32, a);
#pragma unroll
    for (int t = 0; t < 4; ++t) {
      Frag b;
      load_frag(Bt, 16 * t + (lane & 15), 32, 0, b);
      acc[t] = wmma_bf16(a, b, acc[t]);
    }
    if (wave == 0) __builtin_amdgcn_s_wait_tensorcnt(0);
    __syncthreads();
    lds_written_fence(smem);
  }

#pragma unroll
  for (int t = 0; t < 4; ++t) {
    const int col = colbase + 16 * t + (lane & 15);
    const float bi = bias[col];
#pragma unroll
    for (int j = 0; j < 8; ++j) {
      const int token = rowbase + 8 * half + j;
      float val = acc[t][j] + bi;
      if (z == 0) val *= 0.125f;                // fold 1/sqrt(DH) into Q
      if (z < 2) {
        uint16_t* out = (z == 0) ? Qb : Kb;
        out[(size_t)token * D_MODEL + col] = f2bf(val);
      } else {
        const int bb = token >> 11, ss = token & (S_LEN - 1);
        const int hh = col >> 6,  dd = col & (DHEAD - 1);
        Vt[(((size_t)bb * NH + hh) * DHEAD + dd) * S_LEN + ss] = f2bf(val);
      }
    }
  }
}

// ----------------------------------------------------------- attention ----
// grid = 768 (= B*H*16), block = 256 (8 waves); wave handles 16 queries.
// K tile (32 keys x 64 d) and V^T tile (64 d x 32 keys) TDM-staged in LDS,
// double-buffered, software-pipelined.
// LDS map: K buf i at i*4096 ([key][dim], ld=64); V buf i at 8192+i*4096
// ([dim][key], ld=32).
__global__ __launch_bounds__(256)
void ra_attn(const uint16_t* __restrict__ Qb,
             const uint16_t* __restrict__ Kb,
             const uint16_t* __restrict__ Vt,
             const unsigned char* __restrict__ mask,
             uint16_t* __restrict__ Ctx) {
  __shared__ __align__(16) char smem[16384];

  const int tid  = threadIdx.x;
  const int lane = tid & 31;
  const int wave = tid >> 5;
  const int half = lane >> 4;

  const int qblk = blockIdx.x & 15;
  const int bh   = blockIdx.x >> 4;
  const int b    = bh / NH;
  const int h    = bh % NH;
  const int q    = qblk * 128 + wave * 16 + (lane & 15);

  // Q^T B-fragments for this wave's 16 queries (loop invariant)
  Frag bq0, bq1;
  load_frag(Qb, b * S_LEN + q, D_MODEL, h * DHEAD + 0,  bq0);
  load_frag(Qb, b * S_LEN + q, D_MODEL, h * DHEAD + 32, bq1);

  const unsigned char* mrow = mask + (size_t)q * S_LEN;
  const uint16_t* Kg = Kb + (size_t)(b * S_LEN) * D_MODEL + h * DHEAD;
  const uint16_t* Vh = Vt + ((size_t)(b * NH + h) * DHEAD) * S_LEN;

  if (wave == 0) {                              // stage chunk 0 -> buf 0
    tdm_load_2d(0u,    Kg, 64, 32, D_MODEL);    // K: 32 rows of 64 dims
    tdm_load_2d(8192u, Vh, 32, 64, S_LEN);      // V^T: 64 rows of 32 keys
    __builtin_amdgcn_s_wait_tensorcnt(0);
  }
  __syncthreads();
  lds_written_fence(smem);

  float m = -1e30f, l = 0.0f;
  v8f acc[4] = {v8f{}, v8f{}, v8f{}, v8f{}};

  const int NC = S_LEN / 32;                    // 64 chunks
  for (int ci = 0; ci < NC; ++ci) {
    const int c = ci * 32;
    const int cur = ci & 1;
    if (wave == 0 && ci + 1 < NC) {             // DMA next chunk -> other buf
      const uint32_t nb = (uint32_t)((cur ^ 1) * 4096);
      tdm_load_2d(nb,         Kg + (size_t)(c + 32) * D_MODEL, 64, 32, D_MODEL);
      tdm_load_2d(8192u + nb, Vh + (c + 32),                   32, 64, S_LEN);
    }

    const uint16_t* Kt = (const uint16_t*)(smem + cur * 4096);
    const uint16_t* Vs = (const uint16_t*)(smem + 8192 + cur * 4096);

    // scores transposed: S^T(16 keys x 16 q) = K_tile(16x64) x Q^T(64x16)
    v8f s0 = v8f{}, s1 = v8f{};
    {
      Frag a;
      load_frag(Kt, (lane & 15),      64, 0,  a);
      s0 = wmma_bf16(a, bq0, s0);
      load_frag(Kt, (lane & 15),      64, 32, a);
      s0 = wmma_bf16(a, bq1, s0);
      load_frag(Kt, 16 + (lane & 15), 64, 0,  a);
      s1 = wmma_bf16(a, bq0, s1);
      load_frag(Kt, 16 + (lane & 15), 64, 32, a);
      s1 = wmma_bf16(a, bq1, s1);
    }

    // mask (two aligned u64 loads -> 16 bool bytes) + online softmax
    const unsigned long long mw0 =
        *(const unsigned long long*)(mrow + c + 8 * half);
    const unsigned long long mw1 =
        *(const unsigned long long*)(mrow + c + 16 + 8 * half);
    float sv[16];
    float cmax = -1e30f;
#pragma unroll
    for (int j = 0; j < 8; ++j) {
      float v = ((mw0 >> (8 * j)) & 1ull) ? s0[j] : -1e9f;
      sv[j] = v; cmax = fmaxf(cmax, v);
    }
#pragma unroll
    for (int j = 0; j < 8; ++j) {
      float v = ((mw1 >> (8 * j)) & 1ull) ? s1[j] : -1e9f;
      sv[8 + j] = v; cmax = fmaxf(cmax, v);
    }
    cmax = fmaxf(cmax, __shfl_xor(cmax, 16, 32));   // merge key halves
    const float mnew  = fmaxf(m, cmax);
    const float scale = __expf(m - mnew);
    float lc = 0.0f;
#pragma unroll
    for (int i = 0; i < 16; ++i) { sv[i] = __expf(sv[i] - mnew); lc += sv[i]; }
    lc += __shfl_xor(lc, 16, 32);
    l = l * scale + lc;
    m = mnew;
#pragma unroll
    for (int t = 0; t < 4; ++t)
#pragma unroll
      for (int j = 0; j < 8; ++j) acc[t][j] *= scale;

    // S^T C-layout IS the B-fragment layout of P^T: just pack bf16 pairs.
    Frag pb;
#pragma unroll
    for (int i = 0; i < 4; ++i) pb.u[i]     = pack2(sv[2 * i],     sv[2 * i + 1]);
#pragma unroll
    for (int i = 0; i < 4; ++i) pb.u[4 + i] = pack2(sv[8 + 2 * i], sv[8 + 2 * i + 1]);

    // ctx^T(16 dims x 16 q) += V^T_tile(16x32) x P^T(32x16)
#pragma unroll
    for (int t = 0; t < 4; ++t) {
      Frag av;
      load_frag(Vs, 16 * t + (lane & 15), 32, 0, av);
      acc[t] = wmma_bf16(av, pb, acc[t]);
    }

    if (wave == 0) __builtin_amdgcn_s_wait_tensorcnt(0);
    __syncthreads();
    lds_written_fence(smem);
  }

  const float inv = 1.0f / l;
  uint16_t* crow = Ctx + (size_t)(b * S_LEN + q) * D_MODEL + h * DHEAD;
#pragma unroll
  for (int t = 0; t < 4; ++t) {
    uint32_t* o = (uint32_t*)(crow + 16 * t + 8 * half);
#pragma unroll
    for (int jj = 0; jj < 4; ++jj)
      o[jj] = pack2(acc[t][2 * jj] * inv, acc[t][2 * jj + 1] * inv);
  }
}

// --------------------------------------------------------- output proj ----
// Same TDM-staged GEMM structure as ra_qkv_gemm, f32 output.
__global__ __launch_bounds__(256)
void ra_oproj(const uint16_t* __restrict__ Ctxb,
              const uint16_t* __restrict__ Wot,
              const float* __restrict__ bo,
              float* __restrict__ Out) {
  __shared__ __align__(16) char smem[8192];

  const int tid  = threadIdx.x;
  const int lane = tid & 31;
  const int wave = tid >> 5;
  const int half = lane >> 4;
  const int rowbase = (blockIdx.x * 8 + wave) * 16;
  const int colbase = blockIdx.y * 64;
  const int arow = rowbase + (lane & 15);

  const uint16_t* wbase = Wot + (size_t)colbase * D_MODEL;

  if (wave == 0) {
    tdm_load_2d(0u, wbase, 32, 64, D_MODEL);
    __builtin_amdgcn_s_wait_tensorcnt(0);
  }
  __syncthreads();
  lds_written_fence(smem);

  v8f acc[4] = {v8f{}, v8f{}, v8f{}, v8f{}};
  const int NK = D_MODEL / 32;
  for (int ki = 0; ki < NK; ++ki) {
    const int cur = ki & 1;
    if (wave == 0 && ki + 1 < NK)
      tdm_load_2d((uint32_t)((cur ^ 1) * 4096), wbase + (ki + 1) * 32,
                  32, 64, D_MODEL);

    const uint16_t* Bt = (const uint16_t*)(smem + cur * 4096);
    Frag a;
    load_frag(Ctxb, arow, D_MODEL, ki * 32, a);
#pragma unroll
    for (int t = 0; t < 4; ++t) {
      Frag b;
      load_frag(Bt, 16 * t + (lane & 15), 32, 0, b);
      acc[t] = wmma_bf16(a, b, acc[t]);
    }
    if (wave == 0) __builtin_amdgcn_s_wait_tensorcnt(0);
    __syncthreads();
    lds_written_fence(smem);
  }

#pragma unroll
  for (int t = 0; t < 4; ++t) {
    const int col = colbase + 16 * t + (lane & 15);
    const float bi = bo[col];
#pragma unroll
    for (int j = 0; j < 8; ++j) {
      const int token = rowbase + 8 * half + j;
      Out[(size_t)token * D_MODEL + col] = acc[t][j] + bi;
    }
  }
}

// -------------------------------------------------------------- launch ----
extern "C" void kernel_launch(void* const* d_in, const int* in_sizes, int n_in,
                              void* d_out, int out_size, void* d_ws, size_t ws_size,
                              hipStream_t stream) {
  (void)in_sizes; (void)n_in; (void)out_size; (void)ws_size;
  const float* X  = (const float*)d_in[0];
  const float* Wq = (const float*)d_in[1];
  const float* bq = (const float*)d_in[2];
  const float* Wk = (const float*)d_in[3];
  const float* bk = (const float*)d_in[4];
  const float* Wv = (const float*)d_in[5];
  const float* bv = (const float*)d_in[6];
  const float* Wo = (const float*)d_in[7];
  const float* bo = (const float*)d_in[8];
  const unsigned char* mask = (const unsigned char*)d_in[9];
  float* Out = (float*)d_out;

  const size_t tokens = (size_t)NB * S_LEN;          // 8192
  const size_t xsz    = tokens * D_MODEL;            // 6291456 elems
  const size_t wsz    = (size_t)D_MODEL * D_MODEL;   // 589824 elems

  char* p = (char*)d_ws;
  uint16_t* Xb   = (uint16_t*)p; p += xsz * 2;
  uint16_t* Wqt  = (uint16_t*)p; p += wsz * 2;
  uint16_t* Wkt  = (uint16_t*)p; p += wsz * 2;
  uint16_t* Wvt  = (uint16_t*)p; p += wsz * 2;
  uint16_t* Wot  = (uint16_t*)p; p += wsz * 2;
  uint16_t* Qb   = (uint16_t*)p; p += xsz * 2;
  uint16_t* Kb   = (uint16_t*)p; p += xsz * 2;
  uint16_t* Vt   = (uint16_t*)p; p += xsz * 2;
  uint16_t* Ctxb = (uint16_t*)p; p += xsz * 2;

  ra_cvt_bf16<<<(int)((xsz + 255) / 256), 256, 0, stream>>>(X, Xb, (int)xsz);
  ra_transpose_w<<<(int)((wsz + 255) / 256), 256, 0, stream>>>(Wq, Wqt);
  ra_transpose_w<<<(int)((wsz + 255) / 256), 256, 0, stream>>>(Wk, Wkt);
  ra_transpose_w<<<(int)((wsz + 255) / 256), 256, 0, stream>>>(Wv, Wvt);
  ra_transpose_w<<<(int)((wsz + 255) / 256), 256, 0, stream>>>(Wo, Wot);

  ra_qkv_gemm<<<dim3(64, 12, 3), 256, 0, stream>>>(Xb, Wqt, Wkt, Wvt,
                                                   bq, bk, bv, Qb, Kb, Vt);
  ra_attn<<<dim3(NB * NH * 16), 256, 0, stream>>>(Qb, Kb, Vt, mask, Ctxb);
  ra_oproj<<<dim3(64, 12), 256, 0, stream>>>(Ctxb, Wot, bo, Out);
}
